// TransitionKernel_62697932587080
// MI455X (gfx1250) — matrix operand order repaired and checked by reading, not compiled, not run on hardware
//
#include <hip/hip_runtime.h>

typedef __attribute__((ext_vector_type(16))) __bf16 v16bf;
typedef __attribute__((ext_vector_type(8)))  __bf16 v8bf;
typedef __attribute__((ext_vector_type(8)))  float  v8f;
typedef __attribute__((ext_vector_type(4)))  unsigned int u32x4;
typedef __attribute__((ext_vector_type(4)))  int i32x4;
typedef __attribute__((ext_vector_type(8)))  int i32x8;

#define OBS_D 512
#define ACT_D 64
#define HALF_D 256
#define FC_D 512
#define KH 128
#define BATCH 16384

// LDS B-slab row: 32 bf16 data + 8 bf16 pad = 80 bytes. Keeps 16B alignment
// for ds_load_b128 and breaks the 4-way bank conflict of a 64B row stride.
#define LDS_ROW_BF 40
#define SLAB_BF (128 * LDS_ROW_BF)  // one slab: 128 rows

#if __has_builtin(__builtin_amdgcn_tensor_load_to_lds)
#define USE_TDM 1
#else
#define USE_TDM 0
#endif

// ---------------- fp32 -> bf16 conversion ----------------
__global__ void f32_to_bf16_kernel(const float* __restrict__ in,
                                   __bf16* __restrict__ out, int n) {
  int i = blockIdx.x * blockDim.x + threadIdx.x;
  int stride = gridDim.x * blockDim.x;
  for (; i < n; i += stride) out[i] = (__bf16)in[i];
}

// ---------------- A fragment loader (wave32 16x32 bf16 layout) ----------
// lanes 0-15: row=lane, K = kt+[0..15]
// lanes 16-31: row=lane-16, K = kt+[16..31]
__device__ inline v16bf load_a_frag(const __bf16* __restrict__ base, int ld,
                                    int rowBase, int kt, int lane) {
  int m = lane & 15;
  int hi = lane >> 4;
  const __bf16* p = base + (size_t)(rowBase + m) * ld + kt;
  union { v16bf v; v8bf h[2]; } u;
  u.h[0] = *(const v8bf*)(p + (hi ? 16 : 0));
  u.h[1] = *(const v8bf*)(p + (hi ? 24 : 8));
  return u.v;
}

// B fragment from a padded LDS slab: lane n (mod 16) = column, 16
// contiguous K values; upper half-wave offset by 16 K.
__device__ inline v16bf lds_b_frag(const __bf16* buf, int t, int n, int k0) {
  const __bf16* p = buf + (16 * t + n) * LDS_ROW_BF + k0;
  union { v16bf v; v8bf h[2]; } u;
  u.h[0] = *(const v8bf*)(p);
  u.h[1] = *(const v8bf*)(p + 8);
  return u.v;
}

#if USE_TDM
// ---------------- TDM: load one 128-row x 32-elem bf16 slab into LDS -----
// D# per CDNA5 ISA ch.8: group0 = {count|flags, lds_addr, global_addr, type},
// group1 = {mask/data_size/pad, tensor dims, tile dims, strides}.
// Padding: pad_interval=3 (16 dwords = one 64B row), pad_amount=3 (4 dwords)
// -> LDS row stride 80B, matching LDS_ROW_BF.
__device__ inline void tdm_load_b_slab(const __bf16* gsrc, unsigned ldsOff,
                                       unsigned K) {
  unsigned long long ga = (unsigned long long)(size_t)gsrc;
  u32x4 g0;
  g0.x = 1u;                                   // count=1, user descriptor
  g0.y = ldsOff;                               // lds_addr (bytes)
  g0.z = (unsigned)ga;                         // global_addr[31:0]
  g0.w = (unsigned)(ga >> 32) | (2u << 30);    // global_addr[56:32] | type=2

  unsigned td0 = K;          // tensor_dim0: elements per row
  unsigned td1 = 1u << 20;   // tensor_dim1: large (tile never OOB)
  i32x8 g1;
  g1[0] = (int)((1u << 16)            // data_size = 1 -> 2 bytes
                | (1u << 20)          // pad_enable
                | (3u << 22)          // pad_interval: 16 dwords
                | (3u << 25));        // pad_amount: 4 dwords
  g1[1] = (int)((td0 & 0xFFFFu) << 16);
  g1[2] = (int)((td0 >> 16) | ((td1 & 0xFFFFu) << 16));
  g1[3] = (int)((td1 >> 16) | (32u << 16));    // tile_dim0 = 32 (K elems)
  g1[4] = (int)(128u);                         // tile_dim1 = 128 rows
  g1[5] = (int)K;                              // tensor_dim0_stride lo
  g1[6] = 0;                                   // stride hi | dim1_stride lo
  g1[7] = 0;

  i32x4 z4 = {0, 0, 0, 0};
#if __clang_major__ >= 23
  i32x8 z8 = {0, 0, 0, 0, 0, 0, 0, 0};
  __builtin_amdgcn_tensor_load_to_lds(g0, g1, z4, z4, z8, 0);
#else
  __builtin_amdgcn_tensor_load_to_lds(g0, g1, z4, z4, 0);
#endif
}
#endif  // USE_TDM

// ---------------- shared GEMM core: acc[8] += A(16 rows) @ W^T(128 cols) --
// K is compile-time: slab loop fully unrolls (no aCur/aNext copies, TDM
// descriptors constant-fold). Per slab, a rotating 3-deep B-fragment window
// (load t+2 while multiplying t) forces distinct live registers so the
// scheduler emits partial s_wait_dscnt instead of a full drain per WMMA.
template <int K>
__device__ inline void gemm_core(const __bf16* __restrict__ A,
                                 const __bf16* __restrict__ W,
                                 int rowBase, int colBase, int lane, int wave,
                                 __bf16* sB, v8f acc[8]) {
  constexpr int nslab = K >> 5;
#if USE_TDM
  const unsigned sBoff = (unsigned)(size_t)sB;  // LDS byte address
  const unsigned bufBytes = (unsigned)(SLAB_BF * sizeof(__bf16));
  if (wave == 0)
    tdm_load_b_slab(W + (size_t)colBase * K, sBoff, (unsigned)K);
#endif

  v16bf aCur = load_a_frag(A, K, rowBase, 0, lane);
  const int n = lane & 15;
  const int k0 = (lane >> 4) << 4;

#pragma unroll
  for (int i = 0; i < nslab; ++i) {
    __bf16* buf = sB + (size_t)(i & 1) * SLAB_BF;
#if USE_TDM
    if (wave == 0) {
      if (i + 1 < nslab) {
        tdm_load_b_slab(W + (size_t)colBase * K + ((i + 1) << 5),
                        sBoff + (unsigned)((i + 1) & 1) * bufBytes,
                        (unsigned)K);
        __builtin_amdgcn_s_wait_tensorcnt((short)1);  // slab i landed
      } else {
        __builtin_amdgcn_s_wait_tensorcnt((short)0);
      }
    }
#else
    {  // cooperative fallback copy: 512 chunks of 16B, 256 threads
      int tid = wave * 32 + lane;
      for (int c = tid; c < 512; c += 256) {
        int row = c >> 2, ko = (c & 3) * 8;
        *(v8bf*)(buf + row * LDS_ROW_BF + ko) =
            *(const v8bf*)(W + (size_t)(colBase + row) * K + (i << 5) + ko);
      }
    }
#endif
    __syncthreads();  // slab i visible to all waves

    v16bf aNext = aCur;
    if (i + 1 < nslab) aNext = load_a_frag(A, K, rowBase, (i + 1) << 5, lane);

    // rotating 3-deep fragment pipeline over the 8 column tiles
    v16bf b0 = lds_b_frag(buf, 0, n, k0);
    v16bf b1 = lds_b_frag(buf, 1, n, k0);
#pragma unroll
    for (int t = 0; t < 8; ++t) {
      v16bf bn = b0;
      if (t + 2 < 8) bn = lds_b_frag(buf, t + 2, n, k0);
      acc[t] = __builtin_amdgcn_wmma_f32_16x16x32_bf16(
          false, aCur, false, b0, (short)0, acc[t], false, false);
      b0 = b1;
      b1 = bn;
    }

    aCur = aNext;
    __syncthreads();  // all waves done with slab i before DMA overwrites
  }
}

// ---------------- generic GEMM: out = [relu](A @ W^T + bias), bf16 out ----
template <int K, int RELU>
__global__ __launch_bounds__(256)
void gemm_bias_bf16_kernel(const __bf16* __restrict__ A,
                           const __bf16* __restrict__ W,
                           const float* __restrict__ bias,
                           __bf16* __restrict__ out,
                           int ldOut, int colOff) {
  __shared__ __bf16 sB[2 * SLAB_BF];
  int wave = threadIdx.x >> 5;
  int lane = threadIdx.x & 31;
  int rowBase = blockIdx.x * 128 + wave * 16;
  int colBase = blockIdx.y * 128;

  v8f acc[8] = {};
  gemm_core<K>(A, W, rowBase, colBase, lane, wave, sB, acc);

  int colLocal = lane & 15;
  int rowOff = (lane >> 4) * 8;
#pragma unroll
  for (int t = 0; t < 8; ++t) {
    float bv = bias[colBase + 16 * t + colLocal];
#pragma unroll
    for (int r = 0; r < 8; ++r) {
      float v = acc[t][r] + bv;
      if (RELU) v = fmaxf(v, 0.0f);
      size_t row = (size_t)(rowBase + rowOff + r);
      out[row * ldOut + colOff + colBase + 16 * t + colLocal] = (__bf16)v;
    }
  }
}

// ---------------- head GEMM + fused bias + softmax ----------------
__global__ __launch_bounds__(256)
void head_softmax_kernel(const __bf16* __restrict__ X,
                         const __bf16* __restrict__ Wk,
                         const float* __restrict__ bk,
                         float* __restrict__ out) {
  __shared__ __bf16 sB[2 * SLAB_BF];
  int wave = threadIdx.x >> 5;
  int lane = threadIdx.x & 31;
  int rowBase = blockIdx.x * 128 + wave * 16;
  int head = blockIdx.y;
  int colBase = head * KH;

  v8f acc[8] = {};
  gemm_core<FC_D>(X, Wk, rowBase, colBase, lane, wave, sB, acc);

  int colLocal = lane & 15;
  int rowOff = (lane >> 4) * 8;
#pragma unroll
  for (int t = 0; t < 8; ++t) {
    float bv = bk[colBase + 16 * t + colLocal];
#pragma unroll
    for (int r = 0; r < 8; ++r) acc[t][r] += bv;
  }

#pragma unroll
  for (int r = 0; r < 8; ++r) {
    // 128 logits of a row live in acc[0..7][r] across a 16-lane group;
    // xor masks 1,2,4,8 stay inside the group on wave32.
    float m = acc[0][r];
#pragma unroll
    for (int t = 1; t < 8; ++t) m = fmaxf(m, acc[t][r]);
#pragma unroll
    for (int mask = 1; mask < 16; mask <<= 1)
      m = fmaxf(m, __shfl_xor(m, mask, 32));

    float e[8];
    float s = 0.0f;
#pragma unroll
    for (int t = 0; t < 8; ++t) { e[t] = __expf(acc[t][r] - m); s += e[t]; }
#pragma unroll
    for (int mask = 1; mask < 16; mask <<= 1) s += __shfl_xor(s, mask, 32);

    float inv = 1.0f / s;
    size_t row = (size_t)(rowBase + rowOff + r);
    size_t base = row * (size_t)(KH * KH) + (size_t)head * KH + colLocal;
#pragma unroll
    for (int t = 0; t < 8; ++t) out[base + 16 * t] = e[t] * inv;
  }
}

// ---------------- launch ----------------
extern "C" void kernel_launch(void* const* d_in, const int* in_sizes, int n_in,
                              void* d_out, int out_size, void* d_ws, size_t ws_size,
                              hipStream_t stream) {
  (void)in_sizes; (void)n_in; (void)out_size; (void)ws_size;
  const float* obs    = (const float*)d_in[0];
  const float* action = (const float*)d_in[1];
  const float* Wo     = (const float*)d_in[2];
  const float* bo     = (const float*)d_in[3];
  const float* Wa     = (const float*)d_in[4];
  const float* ba     = (const float*)d_in[5];
  const float* Wfc    = (const float*)d_in[6];
  const float* bfc    = (const float*)d_in[7];
  const float* Wk     = (const float*)d_in[8];
  const float* bk     = (const float*)d_in[9];
  float* out = (float*)d_out;

  char* ws = (char*)d_ws;
  auto alloc = [&](size_t elems) {
    char* p = ws;
    ws += (elems * sizeof(__bf16) + 255) & ~(size_t)255;
    return (__bf16*)p;
  };
  __bf16* obs_b = alloc((size_t)BATCH * OBS_D);
  __bf16* act_b = alloc((size_t)BATCH * ACT_D);
  __bf16* Wo_b  = alloc((size_t)HALF_D * OBS_D);
  __bf16* Wa_b  = alloc((size_t)HALF_D * ACT_D);
  __bf16* Wfc_b = alloc((size_t)3 * FC_D * FC_D);
  __bf16* Wk_b  = alloc((size_t)KH * KH * FC_D);
  __bf16* xA    = alloc((size_t)BATCH * FC_D);
  __bf16* xB    = alloc((size_t)BATCH * FC_D);

  auto cvt = [&](const float* src, __bf16* dst, int n) {
    int blocks = (n + 1023) / 1024;
    if (blocks > 4096) blocks = 4096;
    f32_to_bf16_kernel<<<blocks, 256, 0, stream>>>(src, dst, n);
  };
  cvt(obs,    obs_b, BATCH * OBS_D);
  cvt(action, act_b, BATCH * ACT_D);
  cvt(Wo,     Wo_b,  HALF_D * OBS_D);
  cvt(Wa,     Wa_b,  HALF_D * ACT_D);
  cvt(Wfc,    Wfc_b, 3 * FC_D * FC_D);
  cvt(Wk,     Wk_b,  KH * KH * FC_D);

  dim3 blk(256);
  // obs projection -> cols [0,256) of xA, fused relu (relu precedes fc0)
  gemm_bias_bf16_kernel<OBS_D, 1>
      <<<dim3(BATCH / 128, HALF_D / 128), blk, 0, stream>>>(
          obs_b, Wo_b, bo, xA, FC_D, 0);
  // action projection -> cols [256,512) of xA, fused relu
  gemm_bias_bf16_kernel<ACT_D, 1>
      <<<dim3(BATCH / 128, HALF_D / 128), blk, 0, stream>>>(
          act_b, Wa_b, ba, xA, FC_D, HALF_D);
  // fc0 / fc1 with fused relu-for-next-layer
  gemm_bias_bf16_kernel<FC_D, 1>
      <<<dim3(BATCH / 128, FC_D / 128), blk, 0, stream>>>(
          xA, Wfc_b + 0 * FC_D * FC_D, bfc + 0 * FC_D, xB, FC_D, 0);
  gemm_bias_bf16_kernel<FC_D, 1>
      <<<dim3(BATCH / 128, FC_D / 128), blk, 0, stream>>>(
          xB, Wfc_b + 1 * FC_D * FC_D, bfc + 1 * FC_D, xA, FC_D, 0);
  // fc2 output feeds heads directly: NO relu
  gemm_bias_bf16_kernel<FC_D, 0>
      <<<dim3(BATCH / 128, FC_D / 128), blk, 0, stream>>>(
          xA, Wfc_b + 2 * FC_D * FC_D, bfc + 2 * FC_D, xB, FC_D, 0);
  // heads + fused softmax, fp32 out
  head_softmax_kernel<<<dim3(BATCH / 128, KH), blk, 0, stream>>>(
      xB, Wk_b, bk, out);
}
